// MultiHeadAttention_62766652064063
// MI455X (gfx1250) — matrix operand order repaired and checked by reading, not compiled, not run on hardware
//
#include <hip/hip_runtime.h>
#include <hip/hip_bf16.h>

// ---------------------------------------------------------------------------
// MHA with projections for MI455X (gfx1250, wave32, WMMA bf16->f32).
// All GEMM work (4 projections + QK^T + PV) goes through
// v_wmma_f32_16x16x32_bf16. Attention computes scores^T = K*Q^T so that
// every WMMA fragment is a contiguous global load and softmax row stats
// reduce with a single shfl_xor(16).
//
// Round 3: explicit ping-pong double buffering (2x-unrolled streaming loops)
// so the compiler emits no kan->ka register-copy ladder and each
// global_load_b128 is waited on only by the WMMA that consumes it one
// iteration later. global_prefetch_b8 two chunks ahead for K/V streams.
//
// Workspace layout (needs 40 MB):
//   [ 0MB) WqT bf16 1024x1024   [ 2MB) WkT   [ 4MB) WvT   [ 6MB) WoT
//   [ 8MB) Qh  bf16 [B,H,Nq,64]
//   [16MB) Kh  bf16 [B,H,S,64]
//   [24MB) VTh bf16 [B,H,64,S]      (V stored transposed)
//   [32MB) merged bf16 [B,Nq,1024]
// ---------------------------------------------------------------------------

typedef __attribute__((ext_vector_type(16))) __bf16 v16bf;
typedef __attribute__((ext_vector_type(8)))  __bf16 v8bf;
typedef __attribute__((ext_vector_type(8)))  float  v8f;

#define B_  2
#define NQ  2048
#define SQ  2048
#define D_  1024
#define H_  16
#define DH  64

__device__ inline v16bf cat8(v8bf lo, v8bf hi) {
  return __builtin_shufflevector(lo, hi, 0,1,2,3,4,5,6,7,8,9,10,11,12,13,14,15);
}

// A-fragment (16x32, 16-bit): lanes 0-15 row m, VGPR0-3 K=0..7, VGPR4-7 K=16..23;
// lanes 16-31 get K=8..15 / 24..31.  -> two contiguous 16B chunks per lane.
__device__ inline v16bf load_afrag_bf16(const __bf16* row, int kb, int lg) {
  v8bf lo = *(const v8bf*)(row + kb + lg * 8);
  v8bf hi = *(const v8bf*)(row + kb + 16 + lg * 8);
  return cat8(lo, hi);
}

__device__ inline v16bf load_afrag_f32(const float* row, int kb, int lg) {
  const float* p0 = row + kb + lg * 8;
  const float* p1 = row + kb + 16 + lg * 8;
  float4 a0 = *(const float4*)(p0);
  float4 a1 = *(const float4*)(p0 + 4);
  float4 b0 = *(const float4*)(p1);
  float4 b1 = *(const float4*)(p1 + 4);
  v16bf r;
  r[0]=(__bf16)a0.x; r[1]=(__bf16)a0.y; r[2]=(__bf16)a0.z; r[3]=(__bf16)a0.w;
  r[4]=(__bf16)a1.x; r[5]=(__bf16)a1.y; r[6]=(__bf16)a1.z; r[7]=(__bf16)a1.w;
  r[8]=(__bf16)b0.x; r[9]=(__bf16)b0.y; r[10]=(__bf16)b0.z; r[11]=(__bf16)b0.w;
  r[12]=(__bf16)b1.x; r[13]=(__bf16)b1.y; r[14]=(__bf16)b1.z; r[15]=(__bf16)b1.w;
  return r;
}

// B-fragment (32x16, 16-bit): lane holds column n = lane%16; lanes 0-15 K=0..7,16..23,
// lanes 16-31 K=8..15,24..31. With W stored transposed [N][K] this is two 16B chunks.
__device__ inline v16bf load_bfrag(const __bf16* wt_row, int kb, int lg) {
  v8bf lo = *(const v8bf*)(wt_row + kb + lg * 8);
  v8bf hi = *(const v8bf*)(wt_row + kb + 16 + lg * 8);
  return cat8(lo, hi);
}

__device__ inline v8f wmma_bf16(v16bf a, v16bf b, v8f c) {
  return __builtin_amdgcn_wmma_f32_16x16x32_bf16(
      /*neg_a=*/false, a, /*neg_b=*/false, b,
      /*c_mod=*/(short)0, c, /*reuse_a=*/false, /*reuse_b=*/false);
}

// --------------------------- weight convert+transpose -----------------------
__global__ void wconv_kernel(const float* __restrict__ W, __bf16* __restrict__ WT) {
  const int K = 1024, N = 1024;
  int idx = blockIdx.x * blockDim.x + threadIdx.x;   // over N*K
  int n = idx >> 10;        // idx / K
  int k = idx & (K - 1);
  WT[idx] = (__bf16)W[(size_t)k * N + n];            // WT[n][k] = W[k][n]
}

// --------------------------- generic WMMA GEMM ------------------------------
// C[4096 x 1024] = A[4096 x 1024] * W,  W given transposed bf16 [1024][1024].
// MODE 0: store Q head-major  [B,H,Nq,64]  bf16
// MODE 1: store K head-major  [B,H,S ,64]  bf16
// MODE 2: store V transposed  [B,H,64,S ]  bf16
// MODE 3: store f32 row-major [4096][1024] (final output)
template<typename AT, int MODE>
__global__ __launch_bounds__(256)
void gemm_proj(const AT* __restrict__ A, const __bf16* __restrict__ WT,
               void* __restrict__ out) {
  const int K = 1024;
  int lane = threadIdx.x & 31;
  int wave = threadIdx.x >> 5;          // 8 waves
  int lg   = lane >> 4;
  int ln16 = lane & 15;
  int m0 = blockIdx.x * 64 + (wave >> 1) * 16;   // wave's 16 rows
  int n0 = blockIdx.y * 128 + (wave & 1) * 64;   // wave's 64 cols
  const AT* arow = A + (size_t)(m0 + ln16) * K;

  const __bf16* wrow[4];
#pragma unroll
  for (int nt = 0; nt < 4; ++nt)
    wrow[nt] = WT + (size_t)(n0 + nt * 16 + ln16) * K;

  auto loadA = [&](int kb) -> v16bf {
    if constexpr (sizeof(AT) == 4) return load_afrag_f32((const float*)arow, kb, lg);
    else                           return load_afrag_bf16((const __bf16*)arow, kb, lg);
  };

  // Ping-pong double buffer: no copies, loads for step kb issued at kb-32.
  v16bf af0 = loadA(0), af1;
  v16bf bf0[4], bf1[4];
#pragma unroll
  for (int nt = 0; nt < 4; ++nt) bf0[nt] = load_bfrag(wrow[nt], 0, lg);

  v8f acc[4] = {};
  for (int kb = 0; kb < K; kb += 64) {
    // even step: consume buffer 0, fill buffer 1 for kb+32 (always < K)
    af1 = loadA(kb + 32);
#pragma unroll
    for (int nt = 0; nt < 4; ++nt) bf1[nt] = load_bfrag(wrow[nt], kb + 32, lg);
#pragma unroll
    for (int nt = 0; nt < 4; ++nt) acc[nt] = wmma_bf16(af0, bf0[nt], acc[nt]);

    // odd step: consume buffer 1, fill buffer 0 for kb+64
    if (kb + 64 < K) {
      af0 = loadA(kb + 64);
#pragma unroll
      for (int nt = 0; nt < 4; ++nt) bf0[nt] = load_bfrag(wrow[nt], kb + 64, lg);
    }
#pragma unroll
    for (int nt = 0; nt < 4; ++nt) acc[nt] = wmma_bf16(af1, bf1[nt], acc[nt]);
  }

  // D layout: lane (group lg) VGPR j -> row m0 + j + 8*lg, col = n-tile + lane%16
#pragma unroll
  for (int nt = 0; nt < 4; ++nt) {
    int c = n0 + nt * 16 + ln16;
#pragma unroll
    for (int j = 0; j < 8; ++j) {
      int   r = m0 + j + 8 * lg;
      float v = acc[nt][j];
      if constexpr (MODE == 3) {
        ((float*)out)[(size_t)r * 1024 + c] = v;
      } else {
        int b = r >> 11, s = r & 2047;       // Nq == S == 2048
        int h = c >> 6,  d = c & 63;
        __bf16* o = (__bf16*)out;
        if constexpr (MODE == 2)
          o[((size_t)(b * H_ + h) * DH + d) * SQ + s] = (__bf16)v;   // V^T
        else
          o[((size_t)(b * H_ + h) * NQ + s) * DH + d] = (__bf16)v;   // Q / K
      }
    }
  }
}

// --------------------------- flash attention --------------------------------
// scores^T = K_chunk(16x64) @ Q^T(64x16); online softmax over ALL keys;
// strict-lower mask applied only to the PV weights (faithful to reference);
// Sout^T = V^T(64xS) @ P^T(Sx16). One wave owns 16 queries.
__global__ __launch_bounds__(256)
void attn_kernel(const __bf16* __restrict__ Qh, const __bf16* __restrict__ Kh,
                 const __bf16* __restrict__ VTh, __bf16* __restrict__ merged) {
  int lane = threadIdx.x & 31;
  int wave = threadIdx.x >> 5;          // 8 waves -> 128 queries per block
  int lg   = lane >> 4;
  int ln16 = lane & 15;

  int idx = blockIdx.x;                 // B*H*(NQ/128) = 512 blocks
  int qt  = idx & 15;
  int h   = (idx >> 4) & (H_ - 1);
  int b   = idx >> 8;
  int qw  = qt * 128 + wave * 16;       // this wave's query base
  int qg  = qw + ln16;                  // this lane's query (column of scores^T)

  const __bf16* Qbh = Qh  + (size_t)(b * H_ + h) * NQ * DH;
  const __bf16* Kbh = Kh  + (size_t)(b * H_ + h) * SQ * DH;
  const __bf16* Vbh = VTh + (size_t)(b * H_ + h) * DH * SQ;

  // Q^T B-fragments (K-dim = dh), pre-scaled by 1/sqrt(64)
  const __bf16* qrow = Qbh + (size_t)qg * DH;
  v16bf qf[2];
#pragma unroll
  for (int ks = 0; ks < 2; ++ks) {
    v8bf lo = *(const v8bf*)(qrow + ks * 32 + lg * 8);
    v8bf hi = *(const v8bf*)(qrow + ks * 32 + 16 + lg * 8);
    v16bf t = cat8(lo, hi);
#pragma unroll
    for (int i = 0; i < 16; ++i) t[i] = (__bf16)((float)t[i] * 0.125f);
    qf[ks] = t;
  }

  const __bf16* krow0 = Kbh + (size_t)ln16 * DH;   // lane's K row at s0=0

  auto loadK = [&](v16bf (&ka)[4], int s0) {
    const __bf16* k0row = krow0 + (size_t)s0 * DH;
    const __bf16* k1row = k0row + 16 * DH;
    ka[0] = load_afrag_bf16(k0row, 0,  lg);
    ka[1] = load_afrag_bf16(k0row, 32, lg);
    ka[2] = load_afrag_bf16(k1row, 0,  lg);
    ka[3] = load_afrag_bf16(k1row, 32, lg);
  };

  v8f   acc[4] = {};                    // Sout^T: 4 dh-slices of 16 x 16q
  float m_run = -3.0e38f, l_run = 0.f;

  // One chunk of 32 keys; kc = current K fragments, kn = buffer to fill for
  // the next chunk (ping-pong roles fixed at compile time by 2x unroll).
  auto step = [&](int s0, v16bf (&kc)[4], v16bf (&kn)[4]) {
    bool doPV = (s0 < qw + 15);   // some key in chunk can be < some query

    // prefetch two chunks ahead (global_prefetch_b8)
    if (s0 + 64 < SQ) {
      __builtin_prefetch(krow0 + (size_t)(s0 + 64) * DH, 0, 3);
      __builtin_prefetch(Vbh + (size_t)ln16 * SQ + s0 + 64, 0, 3);
    }

    // issue next chunk's K loads before consuming this chunk
    if (s0 + 32 < SQ) loadK(kn, s0 + 32);

    // scores^T for keys [s0, s0+32)
    v8f t0 = {}, t1 = {};
    t0 = wmma_bf16(kc[0], qf[0], t0);
    t0 = wmma_bf16(kc[1], qf[1], t0);
    t1 = wmma_bf16(kc[2], qf[0], t1);
    t1 = wmma_bf16(kc[3], qf[1], t1);

    // issue V^T fragment loads now; consumed after softmax math
    v16bf vf[4];
    if (doPV) {
#pragma unroll
      for (int t = 0; t < 4; ++t) {
        const __bf16* vrow = Vbh + (size_t)(t * 16 + ln16) * SQ + s0;
        v8bf lo = *(const v8bf*)(vrow + lg * 8);
        v8bf hi = *(const v8bf*)(vrow + 16 + lg * 8);
        vf[t] = cat8(lo, hi);
      }
    }

    // online softmax stats (per query = per lane column)
    float cm = -3.0e38f;
#pragma unroll
    for (int j = 0; j < 8; ++j) { cm = fmaxf(cm, t0[j]); cm = fmaxf(cm, t1[j]); }
    cm = fmaxf(cm, __shfl_xor(cm, 16, 32));
    float m_new = fmaxf(m_run, cm);
    float alpha = __expf(m_run - m_new);
    float p0[8], p1[8], ls = 0.f;
#pragma unroll
    for (int j = 0; j < 8; ++j) {
      p0[j] = __expf(t0[j] - m_new);
      p1[j] = __expf(t1[j] - m_new);
      ls += p0[j] + p1[j];
    }
    ls += __shfl_xor(ls, 16, 32);
    l_run = l_run * alpha + ls;         // denominator over ALL keys
    m_run = m_new;
#pragma unroll
    for (int t = 0; t < 4; ++t)
#pragma unroll
      for (int j = 0; j < 8; ++j) acc[t][j] *= alpha;

    // PV accumulation only where some key can be < some query
    if (doPV) {
      float w0[8], w1[8];
#pragma unroll
      for (int j = 0; j < 8; ++j) {
        int k0g = s0 + j + 8 * lg;      // key index of score row
        w0[j] = (k0g      < qg) ? p0[j] : 0.f;   // strict lower triangle
        w1[j] = (k0g + 16 < qg) ? p1[j] : 0.f;
      }
      // Build P^T B-fragment (rows = keys s0+lg*8+0..7 and s0+16+lg*8+0..7,
      // col = query) directly from own registers -- no LDS round trip.
      v16bf pf;
#pragma unroll
      for (int j = 0; j < 8; ++j) {
        pf[j]     = (__bf16)w0[j];
        pf[8 + j] = (__bf16)w1[j];
      }
#pragma unroll
      for (int t = 0; t < 4; ++t) acc[t] = wmma_bf16(vf[t], pf, acc[t]);
    }
  };

  v16bf kbuf0[4], kbuf1[4];
  loadK(kbuf0, 0);
  for (int s0 = 0; s0 < SQ; s0 += 64) {   // SQ/32 = 64 chunks, 2 per trip
    step(s0,      kbuf0, kbuf1);
    step(s0 + 32, kbuf1, kbuf0);
  }

  float inv = 1.0f / l_run;
#pragma unroll
  for (int t = 0; t < 4; ++t)
#pragma unroll
    for (int j = 0; j < 8; ++j) {
      int dh = t * 16 + j + 8 * lg;
      merged[(size_t)(b * NQ + qg) * (H_ * DH) + h * DH + dh] =
          (__bf16)(acc[t][j] * inv);
    }
}

// ---------------------------------------------------------------------------
extern "C" void kernel_launch(void* const* d_in, const int* in_sizes, int n_in,
                              void* d_out, int out_size, void* d_ws, size_t ws_size,
                              hipStream_t stream) {
  (void)in_sizes; (void)n_in; (void)out_size; (void)ws_size;
  const float* queries = (const float*)d_in[0];
  const float* keys    = (const float*)d_in[1];
  const float* values  = (const float*)d_in[2];
  const float* Wq      = (const float*)d_in[3];
  const float* Wk      = (const float*)d_in[4];
  const float* Wv      = (const float*)d_in[5];
  const float* Wo      = (const float*)d_in[6];

  char* ws = (char*)d_ws;
  __bf16* WqT = (__bf16*)(ws + (size_t)0);
  __bf16* WkT = (__bf16*)(ws + ((size_t)2  << 20));
  __bf16* WvT = (__bf16*)(ws + ((size_t)4  << 20));
  __bf16* WoT = (__bf16*)(ws + ((size_t)6  << 20));
  __bf16* Qh  = (__bf16*)(ws + ((size_t)8  << 20));
  __bf16* Kh  = (__bf16*)(ws + ((size_t)16 << 20));
  __bf16* VTh = (__bf16*)(ws + ((size_t)24 << 20));
  __bf16* Mg  = (__bf16*)(ws + ((size_t)32 << 20));

  const int wn = 1024 * 1024;
  wconv_kernel<<<wn / 256, 256, 0, stream>>>(Wq, WqT);
  wconv_kernel<<<wn / 256, 256, 0, stream>>>(Wk, WkT);
  wconv_kernel<<<wn / 256, 256, 0, stream>>>(Wv, WvT);
  wconv_kernel<<<wn / 256, 256, 0, stream>>>(Wo, WoT);

  dim3 gg(64, 8);   // 4096 rows / 64, 1024 cols / 128
  gemm_proj<float, 0><<<gg, 256, 0, stream>>>(queries, WqT, Qh);
  gemm_proj<float, 1><<<gg, 256, 0, stream>>>(keys,    WkT, Kh);
  gemm_proj<float, 2><<<gg, 256, 0, stream>>>(values,  WvT, VTh);

  attn_kernel<<<B_ * H_ * (NQ / 128), 256, 0, stream>>>(Qh, Kh, VTh, Mg);

  gemm_proj<__bf16, 3><<<gg, 256, 0, stream>>>(Mg, WoT, d_out);
}
